// KozyraRNNWithAttention_48473000902922
// MI455X (gfx1250) — matrix operand order repaired and checked by reading, not compiled, not run on hardware
//
#include <hip/hip_runtime.h>

// ---------------------------------------------------------------------------
// Problem constants (reference: B,T,D,H,NH = 512,256,32,256,4)
// ---------------------------------------------------------------------------
#define Bc 512
#define Tc 256
#define Dc 32
#define Hc 256
#define NHc 4
#define HDc 64
#define K0c 320   // layer0 GEMM K: [x(32) | pd(1) | pad(31) | h1(256)]
#define K1c 512   // layer1 GEMM K: [h1(256) | h2(256)]

typedef __bf16 v16bf __attribute__((ext_vector_type(16)));
typedef float v8f __attribute__((ext_vector_type(8)));
typedef unsigned short v16us __attribute__((ext_vector_type(16)));
typedef unsigned short v8us __attribute__((ext_vector_type(8)));

__device__ __forceinline__ unsigned short f2bf(float f) {
  unsigned int u = __float_as_uint(f);
  u += 0x7FFFu + ((u >> 16) & 1u);   // round-to-nearest-even
  return (unsigned short)(u >> 16);
}

__device__ __forceinline__ float sigf(float x) {
  return 1.0f / (1.0f + __expf(-x));
}
__device__ __forceinline__ float tanh_fast(float x) {
  x = fminf(fmaxf(x, -15.0f), 15.0f);
  float e = __expf(2.0f * x);
  return (e - 1.0f) / (e + 1.0f);
}

// D = A(16x32 bf16) * B(32x16 bf16) + C(16x16 f32), wave32 WMMA
__device__ __forceinline__ v8f wmma_bf16(v16us a, v16us b, v8f c) {
  return __builtin_amdgcn_wmma_f32_16x16x32_bf16(
      false, __builtin_bit_cast(v16bf, a),
      false, __builtin_bit_cast(v16bf, b),
      (short)0, c, false, false);
}

// A fragment: row-major [M x K] bf16 buffer, tile origin p=(row0,k0), stride ld.
// Lane l: m = l&15, half h = l>>4; element j: k = (j>>3)*16 + h*8 + (j&7)
//  -> two contiguous 16B chunks at k = h*8 and k = 16 + h*8.
__device__ __forceinline__ v16us load_a(const unsigned short* p, int ld) {
  const int l = threadIdx.x & 31;
  const int m = l & 15, hh = l >> 4;
  const unsigned short* r0 = p + m * ld + hh * 8;
  v8us lo = *(const v8us*)(r0);
  v8us hi = *(const v8us*)(r0 + 16);
  v16us out;
#pragma unroll
  for (int i = 0; i < 8; ++i) { out[i] = lo[i]; out[i + 8] = hi[i]; }
  return out;
}

// B fragment: "weight-style" row-major [N x K] bf16 buffer (element (n,k) at
// n*ld + k), tile origin p=(n0,k0). Lane l: n = l&15, element j: k = (j>>3)*16 + (l>>4)*8 + (j&7)
//  -> two contiguous 16B chunks at k = h*8 and k = 16 + h*8.
__device__ __forceinline__ v16us load_b(const unsigned short* p, int ld) {
  const int l = threadIdx.x & 31;
  const int n = l & 15, hh = l >> 4;
  const unsigned short* r0 = p + n * ld + hh * 8;
  v8us lo = *(const v8us*)(r0);
  v8us hi = *(const v8us*)(r0 + 16);
  v16us out;
#pragma unroll
  for (int i = 0; i < 8; ++i) { out[i] = lo[i]; out[i + 8] = hi[i]; }
  return out;
}

// Per-lane prefetch of a B-fragment's cacheline (global_prefetch_b8)
__device__ __forceinline__ void prefetch_b(const unsigned short* p, int ld) {
  const int l = threadIdx.x & 31;
  __builtin_prefetch(p + (l & 15) * ld + (l >> 4) * 16, 0, 3);
}

// ---------------------------------------------------------------------------
// Prep kernels: fold/convert weights to bf16 GEMM layouts (all L2-resident)
// ---------------------------------------------------------------------------
__global__ void prep_f2bf(const float* __restrict__ src,
                          unsigned short* __restrict__ dst, int n) {
  int i = blockIdx.x * blockDim.x + threadIdx.x;
  if (i < n) dst[i] = f2bf(src[i]);
}

__global__ void prep_w0cat(const float* __restrict__ Wih0,
                           const float* __restrict__ Whh0,
                           unsigned short* __restrict__ dst) {
  int i = blockIdx.x * blockDim.x + threadIdx.x;  // 1024*320
  if (i >= 1024 * K0c) return;
  int n = i / K0c, k = i % K0c;
  float v = 0.0f;
  if (k < 33) v = Wih0[n * 33 + k];
  else if (k >= 64) v = Whh0[n * 256 + (k - 64)];
  dst[i] = f2bf(v);
}

__global__ void prep_w1cat(const float* __restrict__ Wih1,
                           const float* __restrict__ Whh1,
                           unsigned short* __restrict__ dst) {
  int i = blockIdx.x * blockDim.x + threadIdx.x;  // 1024*512
  if (i >= 1024 * K1c) return;
  int n = i / K1c, k = i % K1c;
  float v = (k < 256) ? Wih1[n * 256 + k] : Whh1[n * 256 + (k - 256)];
  dst[i] = f2bf(v);
}

__global__ void prep_bias(const float* __restrict__ a,
                          const float* __restrict__ b,
                          float* __restrict__ dst, int n) {
  int i = blockIdx.x * blockDim.x + threadIdx.x;
  if (i < n) dst[i] = a[i] + b[i];
}

__global__ void prep_w1split(const float* __restrict__ W1,
                             unsigned short* __restrict__ W1a,
                             unsigned short* __restrict__ W1b,
                             unsigned short* __restrict__ W1eff) {
  int i = blockIdx.x * blockDim.x + threadIdx.x;  // 256*256
  if (i >= 256 * 256) return;
  int n = i >> 8, k = i & 255;
  float a = W1[n * 512 + k];
  float b = W1[n * 512 + 256 + k];
  W1a[i] = f2bf(a);
  W1b[i] = f2bf(b);
  W1eff[i] = f2bf(a + b);  // concat(h2,h2) @ W1.T == h2 @ (W1a+W1b).T
}

// ---------------------------------------------------------------------------
// Recurrent scan: 32 WGs x 16 batch rows, 8 waves. Sequential over T.
// Weights are streamed from L2 every step (blocked from LICM-hoisting by an
// in-loop memory clobber; hoisting 80x8-VGPR fragments would spill to scratch).
// ---------------------------------------------------------------------------
__global__ void __launch_bounds__(256)
lstm_scan(const float* __restrict__ feat,              // [B,T,D] f32
          const unsigned short* __restrict__ W0cat,    // [1024,320] bf16
          const unsigned short* __restrict__ W1cat,    // [1024,512] bf16
          const float* __restrict__ bias0,             // [1024]
          const float* __restrict__ bias1,             // [1024]
          const unsigned short* __restrict__ W1eff,    // [256,256] bf16
          const float* __restrict__ b1,                // [256]
          const float* __restrict__ W2,                // [256]
          const float* __restrict__ b2,                // [1]
          unsigned short* __restrict__ lstm_bf) {      // [B*T,256] bf16
  __shared__ unsigned short A0[16 * K0c];   // [x | pd | pad | h1_{t-1}]
  __shared__ unsigned short A1[16 * K1c];   // [h1_t | h2_{t-1}]
  __shared__ unsigned short Afc[16 * Hc];   // h2_t for feedback fc
  __shared__ float pdbuf[16];

  const int b0 = blockIdx.x * 16;
  const int tid = threadIdx.x;
  const int w = tid >> 5;
  const int l = tid & 31;
  const int lm = l & 15, hh = l >> 4;

  for (int i = tid; i < 16 * K0c; i += 256) A0[i] = 0;
  for (int i = tid; i < 16 * K1c; i += 256) A1[i] = 0;
  for (int i = tid; i < 16 * Hc; i += 256) Afc[i] = 0;
  if (tid < 16) pdbuf[tid] = 0.0f;
  // x_0
  for (int i = tid; i < 16 * Dc; i += 256) {
    int m = i >> 5, d = i & 31;
    A0[m * K0c + d] = f2bf(feat[(b0 + m) * (Tc * Dc) + d]);
  }

  // Manually hoist loop-invariant per-lane scalars into registers (the memory
  // clobber below would otherwise force re-loading them every iteration).
  float bi0[2][4], bi1[2][4], b1r[2], w2r[2];
#pragma unroll
  for (int sub = 0; sub < 2; ++sub) {
    int u = w * 32 + sub * 16 + lm;
#pragma unroll
    for (int g = 0; g < 4; ++g) {
      bi0[sub][g] = bias0[g * Hc + u];
      bi1[sub][g] = bias1[g * Hc + u];
    }
    b1r[sub] = b1[u];
    w2r[sub] = W2[u];
  }
  const float b2v = b2[0];

  v8f c1[2] = {};  // cell state layer0, per-wave 16x32 slice
  v8f c2[2] = {};  // cell state layer1

  for (int t = 0; t < Tc; ++t) {
    // Opaque memory clobber: keeps the weight-fragment loads inside the loop
    // (streamed from L2) instead of being hoisted + spilled to scratch.
    asm volatile("" ::: "memory");
    __syncthreads();  // A0 (x_t, pd_{t-1}, h1_{t-1}) ready

    // ---- layer 0: gates0[16,1024] = A0[16,320] @ W0cat.T ----
    v8f acc[8] = {};
    for (int kt = 0; kt < K0c / 32; ++kt) {
      if (kt + 1 < K0c / 32)
        prefetch_b(W0cat + (w * 32) * K0c + (kt + 1) * 32, K0c);
      v16us a = load_a(A0 + kt * 32, K0c);
#pragma unroll
      for (int nt = 0; nt < 8; ++nt) {
        int n0 = (nt >> 1) * Hc + w * 32 + (nt & 1) * 16;
        acc[nt] = wmma_bf16(a, load_b(W0cat + n0 * K0c + kt * 32, K0c), acc[nt]);
      }
    }
    __syncthreads();  // all waves done reading A0

#pragma unroll
    for (int sub = 0; sub < 2; ++sub) {
      int u = w * 32 + sub * 16 + lm;
#pragma unroll
      for (int r = 0; r < 8; ++r) {
        float iv = sigf(acc[0 + sub][r] + bi0[sub][0]);
        float fv = sigf(acc[2 + sub][r] + bi0[sub][1]);
        float gv = tanh_fast(acc[4 + sub][r] + bi0[sub][2]);
        float ov = sigf(acc[6 + sub][r] + bi0[sub][3]);
        float c = fv * c1[sub][r] + iv * gv;
        c1[sub][r] = c;
        unsigned short hb = f2bf(ov * tanh_fast(c));
        int m = r + 8 * hh;
        A1[m * K1c + u] = hb;        // h1_t for layer1 (this step)
        A0[m * K0c + 64 + u] = hb;   // h1_t for layer0 (next step)
      }
    }
    __syncthreads();  // A1 = [h1_t | h2_{t-1}] ready

    // ---- layer 1: gates1[16,1024] = A1[16,512] @ W1cat.T ----
    v8f acc1[8] = {};
    for (int kt = 0; kt < K1c / 32; ++kt) {
      if (kt + 1 < K1c / 32)
        prefetch_b(W1cat + (w * 32) * K1c + (kt + 1) * 32, K1c);
      v16us a = load_a(A1 + kt * 32, K1c);
#pragma unroll
      for (int nt = 0; nt < 8; ++nt) {
        int n0 = (nt >> 1) * Hc + w * 32 + (nt & 1) * 16;
        acc1[nt] = wmma_bf16(a, load_b(W1cat + n0 * K1c + kt * 32, K1c), acc1[nt]);
      }
    }
    __syncthreads();  // all waves done reading A1 (old h2)

#pragma unroll
    for (int sub = 0; sub < 2; ++sub) {
      int u = w * 32 + sub * 16 + lm;
#pragma unroll
      for (int r = 0; r < 8; ++r) {
        float iv = sigf(acc1[0 + sub][r] + bi1[sub][0]);
        float fv = sigf(acc1[2 + sub][r] + bi1[sub][1]);
        float gv = tanh_fast(acc1[4 + sub][r] + bi1[sub][2]);
        float ov = sigf(acc1[6 + sub][r] + bi1[sub][3]);
        float c = fv * c2[sub][r] + iv * gv;
        c2[sub][r] = c;
        unsigned short hb = f2bf(ov * tanh_fast(c));
        int m = r + 8 * hh;
        A1[m * K1c + Hc + u] = hb;                    // h2_t for next step
        Afc[m * Hc + u] = hb;                         // h2_t for fc
        lstm_bf[((b0 + m) * Tc + t) * Hc + u] = hb;   // sequence output
      }
    }
    __syncthreads();  // Afc ready

    // ---- feedback fc: pd = relu(h2 @ W1eff.T + b1) @ W2.T + b2 ----
    v8f accf[2] = {};
    for (int kt = 0; kt < Hc / 32; ++kt) {
      v16us a = load_a(Afc + kt * 32, Hc);
#pragma unroll
      for (int sub = 0; sub < 2; ++sub) {
        int n0 = w * 32 + sub * 16;
        accf[sub] = wmma_bf16(a, load_b(W1eff + n0 * Hc + kt * 32, Hc), accf[sub]);
      }
    }
    float part[8];
#pragma unroll
    for (int r = 0; r < 8; ++r) part[r] = 0.0f;
#pragma unroll
    for (int sub = 0; sub < 2; ++sub) {
#pragma unroll
      for (int r = 0; r < 8; ++r) {
        float y = fmaxf(accf[sub][r] + b1r[sub], 0.0f);
        part[r] += y * w2r[sub];
      }
    }
#pragma unroll
    for (int r = 0; r < 8; ++r) atomicAdd(&pdbuf[r + 8 * hh], part[r]);
    __syncthreads();

    if (tid < 16) {
      A0[tid * K0c + 32] = f2bf(pdbuf[tid] + b2v);  // pd feedback column
      pdbuf[tid] = 0.0f;
    }
    if (t + 1 < Tc) {
      for (int i = tid; i < 16 * Dc; i += 256) {
        int m = i >> 5, d = i & 31;
        A0[m * K0c + d] = f2bf(feat[(b0 + m) * (Tc * Dc) + (t + 1) * Dc + d]);
      }
    }
  }
}

// ---------------------------------------------------------------------------
// Flash-style attention: one WG per (batch, head). q/k/vT staged in LDS.
// ---------------------------------------------------------------------------
__global__ void __launch_bounds__(256)
attn_kernel(const unsigned short* __restrict__ lstm_bf,  // [B*T,256]
            const unsigned short* __restrict__ Wip,      // [768,256]
            const float* __restrict__ b_in_proj,         // [768]
            unsigned short* __restrict__ ctxbf) {        // [B*T,256]
  __shared__ unsigned short qs[Tc * HDc];       // 32 KB (q pre-scaled 1/8)
  __shared__ unsigned short ks[Tc * HDc];       // 32 KB (row-major [key,64])
  __shared__ unsigned short vT[HDc * Tc];       // 32 KB (transposed [64,key])
  __shared__ unsigned short Pst[8 * 16 * 128];  // 32 KB per-wave P staging

  const int bidx = blockIdx.x >> 2;
  const int hidx = blockIdx.x & 3;
  const int tid = threadIdx.x;
  const int w = tid >> 5;
  const int l = tid & 31;
  const int lm = l & 15, hh = l >> 4;

  // ---- phase 1: q,k,v = X @ W{q,k,v}.T (head slice), X from L2 ----
  for (int job = w; job < 192; job += 8) {
    int s = job / 64;          // 0=q 1=k 2=v
    int rem = job % 64;
    int mi = rem >> 2, ni = rem & 3;
    v8f acc = {};
    for (int kt = 0; kt < Hc / 32; ++kt) {
      v16us a = load_a(lstm_bf + (bidx * Tc + mi * 16) * Hc + kt * 32, Hc);
      v16us bf = load_b(Wip + (s * Hc + hidx * HDc + ni * 16) * Hc + kt * 32, Hc);
      acc = wmma_bf16(a, bf, acc);
    }
    int ncol = ni * 16 + lm;
    float bb = b_in_proj[s * Hc + hidx * HDc + ncol];
#pragma unroll
    for (int r = 0; r < 8; ++r) {
      int m = mi * 16 + r + 8 * hh;
      float val = acc[r] + bb;
      if (s == 0) qs[m * HDc + ncol] = f2bf(val * 0.125f);  // 1/sqrt(64)
      else if (s == 1) ks[m * HDc + ncol] = f2bf(val);
      else vT[ncol * Tc + m] = f2bf(val);
    }
  }
  __syncthreads();

  // ---- phase 2: per wave, 2 q-tiles of 16 rows ----
  unsigned short* Pw = Pst + w * (16 * 128);
  for (int mi2 = 0; mi2 < 2; ++mi2) {
    int m0 = (w * 2 + mi2) * 16;
    v16us aq0 = load_a(qs + m0 * HDc, HDc);
    v16us aq1 = load_a(qs + m0 * HDc + 32, HDc);

    v8f S[16];
#pragma unroll
    for (int nt = 0; nt < 16; ++nt) {
      v8f acc = {};
      acc = wmma_bf16(aq0, load_b(ks + (nt * 16) * HDc, HDc), acc);
      acc = wmma_bf16(aq1, load_b(ks + (nt * 16) * HDc + 32, HDc), acc);
      S[nt] = acc;
    }

    // row softmax (rows live in 16-lane halves matching C layout)
    float rinv[8];
#pragma unroll
    for (int r = 0; r < 8; ++r) {
      float mx = S[0][r];
#pragma unroll
      for (int nt = 1; nt < 16; ++nt) mx = fmaxf(mx, S[nt][r]);
      for (int msk = 1; msk < 16; msk <<= 1)
        mx = fmaxf(mx, __shfl_xor(mx, msk, 32));
      float sm = 0.0f;
#pragma unroll
      for (int nt = 0; nt < 16; ++nt) {
        float e = __expf(S[nt][r] - mx);
        S[nt][r] = e;
        sm += e;
      }
      for (int msk = 1; msk < 16; msk <<= 1) sm += __shfl_xor(sm, msk, 32);
      rinv[r] = 1.0f / sm;
    }

    // ctx = P @ V, K=256 processed in two 128-wide halves through LDS
    v8f cacc[4] = {};
    for (int half = 0; half < 2; ++half) {
#pragma unroll
      for (int nt8 = 0; nt8 < 8; ++nt8)
#pragma unroll
        for (int r = 0; r < 8; ++r)
          Pw[(r + 8 * hh) * 128 + nt8 * 16 + lm] = f2bf(S[half * 8 + nt8][r]);
      for (int kt = 0; kt < 4; ++kt) {
        v16us a = load_a(Pw + kt * 32, 128);
#pragma unroll
        for (int nt = 0; nt < 4; ++nt) {
          v16us bf = load_b(vT + (nt * 16) * Tc + half * 128 + kt * 32, Tc);
          cacc[nt] = wmma_bf16(a, bf, cacc[nt]);
        }
      }
    }
#pragma unroll
    for (int nt = 0; nt < 4; ++nt)
#pragma unroll
      for (int r = 0; r < 8; ++r) {
        int m = m0 + r + 8 * hh;
        ctxbf[(bidx * Tc + m) * Hc + hidx * HDc + nt * 16 + lm] =
            f2bf(cacc[nt][r] * rinv[r]);
      }
  }
}

// ---------------------------------------------------------------------------
// Out-projection + concat FC head, fused per 16-row tile of (B*T)
// ---------------------------------------------------------------------------
__global__ void __launch_bounds__(256)
final_fc(const unsigned short* __restrict__ lstm_bf,  // [B*T,256]
         const unsigned short* __restrict__ ctxbf,    // [B*T,256]
         const unsigned short* __restrict__ Wop,      // [256,256]
         const float* __restrict__ b_out_proj,        // [256]
         const unsigned short* __restrict__ W1a,      // [256,256]
         const unsigned short* __restrict__ W1b,      // [256,256]
         const float* __restrict__ b1,                // [256]
         const float* __restrict__ W2,                // [256]
         const float* __restrict__ b2,                // [1]
         float* __restrict__ out) {                   // [B*T]
  __shared__ unsigned short attnA[16 * Hc];
  __shared__ float pdbuf[16];

  const int row0 = blockIdx.x * 16;
  const int tid = threadIdx.x;
  const int w = tid >> 5;
  const int l = tid & 31;
  const int lm = l & 15, hh = l >> 4;

  if (tid < 16) pdbuf[tid] = 0.0f;

  // stage 1: attn_out = ctx @ Wop.T + b_out_proj  -> bf16 LDS
  v8f acc[2] = {};
  for (int kt = 0; kt < Hc / 32; ++kt) {
    v16us a = load_a(ctxbf + row0 * Hc + kt * 32, Hc);
#pragma unroll
    for (int sub = 0; sub < 2; ++sub) {
      int n0 = w * 32 + sub * 16;
      acc[sub] = wmma_bf16(a, load_b(Wop + n0 * Hc + kt * 32, Hc), acc[sub]);
    }
  }
#pragma unroll
  for (int sub = 0; sub < 2; ++sub) {
    int n = w * 32 + sub * 16 + lm;
    float bb = b_out_proj[n];
#pragma unroll
    for (int r = 0; r < 8; ++r)
      attnA[(r + 8 * hh) * Hc + n] = f2bf(acc[sub][r] + bb);
  }
  __syncthreads();

  // stage 2: y = relu(lstm@W1a.T + attn@W1b.T + b1); out = y @ W2.T + b2
  v8f acc2[2] = {};
  for (int kt = 0; kt < Hc / 32; ++kt) {
    v16us aL = load_a(lstm_bf + row0 * Hc + kt * 32, Hc);
    v16us aA = load_a(attnA + kt * 32, Hc);
#pragma unroll
    for (int sub = 0; sub < 2; ++sub) {
      int n0 = w * 32 + sub * 16;
      acc2[sub] = wmma_bf16(aL, load_b(W1a + n0 * Hc + kt * 32, Hc), acc2[sub]);
      acc2[sub] = wmma_bf16(aA, load_b(W1b + n0 * Hc + kt * 32, Hc), acc2[sub]);
    }
  }
  float part[8];
#pragma unroll
  for (int r = 0; r < 8; ++r) part[r] = 0.0f;
#pragma unroll
  for (int sub = 0; sub < 2; ++sub) {
    int u = w * 32 + sub * 16 + lm;
    float bb = b1[u], w2v = W2[u];
#pragma unroll
    for (int r = 0; r < 8; ++r) {
      float y = fmaxf(acc2[sub][r] + bb, 0.0f);
      part[r] += y * w2v;
    }
  }
#pragma unroll
  for (int r = 0; r < 8; ++r) atomicAdd(&pdbuf[r + 8 * hh], part[r]);
  __syncthreads();
  if (tid < 16) out[row0 + tid] = pdbuf[tid] + b2[0];
}

// ---------------------------------------------------------------------------
extern "C" void kernel_launch(void* const* d_in, const int* in_sizes, int n_in,
                              void* d_out, int out_size, void* d_ws,
                              size_t ws_size, hipStream_t stream) {
  const float* features = (const float*)d_in[0];
  const float* Wih0 = (const float*)d_in[1];
  const float* Whh0 = (const float*)d_in[2];
  const float* bih0 = (const float*)d_in[3];
  const float* bhh0 = (const float*)d_in[4];
  const float* Wih1 = (const float*)d_in[5];
  const float* Whh1 = (const float*)d_in[6];
  const float* bih1 = (const float*)d_in[7];
  const float* bhh1 = (const float*)d_in[8];
  const float* W_in_proj = (const float*)d_in[9];
  const float* b_in_proj = (const float*)d_in[10];
  const float* W_out_proj = (const float*)d_in[11];
  const float* b_out_proj = (const float*)d_in[12];
  const float* W1 = (const float*)d_in[13];
  const float* b1 = (const float*)d_in[14];
  const float* W2 = (const float*)d_in[15];
  const float* b2 = (const float*)d_in[16];

  char* ws = (char*)d_ws;
  size_t off = 0;
  auto alloc = [&](size_t bytes) -> void* {
    void* p = ws + off;
    off = (off + bytes + 255) & ~(size_t)255;
    return p;
  };
  unsigned short* W0cat = (unsigned short*)alloc(1024 * K0c * 2);
  unsigned short* W1cat = (unsigned short*)alloc(1024 * K1c * 2);
  float* bias0 = (float*)alloc(1024 * 4);
  float* bias1 = (float*)alloc(1024 * 4);
  unsigned short* W1a = (unsigned short*)alloc(256 * 256 * 2);
  unsigned short* W1b = (unsigned short*)alloc(256 * 256 * 2);
  unsigned short* W1eff = (unsigned short*)alloc(256 * 256 * 2);
  unsigned short* Wip = (unsigned short*)alloc(768 * 256 * 2);
  unsigned short* Wop = (unsigned short*)alloc(256 * 256 * 2);
  unsigned short* lstm_bf = (unsigned short*)alloc((size_t)Bc * Tc * Hc * 2);
  unsigned short* ctxbf = (unsigned short*)alloc((size_t)Bc * Tc * Hc * 2);

  dim3 blk(256);
  prep_w0cat<<<(1024 * K0c + 255) / 256, blk, 0, stream>>>(Wih0, Whh0, W0cat);
  prep_w1cat<<<(1024 * K1c + 255) / 256, blk, 0, stream>>>(Wih1, Whh1, W1cat);
  prep_bias<<<4, blk, 0, stream>>>(bih0, bhh0, bias0, 1024);
  prep_bias<<<4, blk, 0, stream>>>(bih1, bhh1, bias1, 1024);
  prep_w1split<<<(256 * 256 + 255) / 256, blk, 0, stream>>>(W1, W1a, W1b, W1eff);
  prep_f2bf<<<(768 * 256 + 255) / 256, blk, 0, stream>>>(W_in_proj, Wip, 768 * 256);
  prep_f2bf<<<(256 * 256 + 255) / 256, blk, 0, stream>>>(W_out_proj, Wop, 256 * 256);

  lstm_scan<<<Bc / 16, blk, 0, stream>>>(features, W0cat, W1cat, bias0, bias1,
                                         W1eff, b1, W2, b2, lstm_bf);
  attn_kernel<<<Bc * NHc, blk, 0, stream>>>(lstm_bf, Wip, b_in_proj, ctxbf);
  final_fc<<<(Bc * Tc) / 16, blk, 0, stream>>>(lstm_bf, ctxbf, Wop, b_out_proj,
                                               W1a, W1b, b1, W2, b2,
                                               (float*)d_out);
}
